// SpatialAligner_30408368456282
// MI455X (gfx1250) — matrix operand order repaired and checked by reading, not compiled, not run on hardware
//
#include <hip/hip_runtime.h>

// ---------------------------------------------------------------------------
// Problem dims (from setup_inputs): B=2, n=16384, m=4096, C1=512, C2=384,
// out_ch=512, k=3.  C = C1+C2 = 896, Ntot = B*n = 32768.
// ---------------------------------------------------------------------------
#define BATCH   2
#define N_TGT   16384
#define M_SRC   4096
#define C1_CH   512
#define C2_CH   384
#define C_IN    896
#define O_CH    512
#define NTOT    32768
#define BN_EPS  1e-5f
#define W_EPS   1e-6f

typedef __attribute__((ext_vector_type(16))) __bf16        v16bf;
typedef __attribute__((ext_vector_type(8)))  float         v8f;
typedef __attribute__((ext_vector_type(4)))  unsigned int  v4u;
typedef __attribute__((ext_vector_type(4)))  float         v4f;

union FragU {         // 8 VGPRs viewed either as 2x b128 loads or a WMMA operand
    v4u   q[2];
    v16bf v;
};

static __device__ __forceinline__ unsigned short f32_to_bf16(float f) {
    unsigned u = __float_as_uint(f);
    u += 0x7FFFu + ((u >> 16) & 1u);   // round-to-nearest-even
    return (unsigned short)(u >> 16);
}

// ---------------------------------------------------------------------------
// Kernel 1: top-3 nearest neighbours + normalized inverse-distance weights.
// One thread per target point; all 4096 src points of the batch live in LDS.
// ---------------------------------------------------------------------------
__global__ void __launch_bounds__(256)
knn_top3(const float* __restrict__ tgt, const float* __restrict__ src,
         int* __restrict__ knn_idx, float* __restrict__ knn_w) {
    __shared__ float sp[M_SRC * 3];                    // 48 KB of 320 KB LDS
    const int t = blockIdx.x * 256 + threadIdx.x;      // global point id
    const int b = t >> 14;                             // t / N_TGT
    const float* sb = src + (size_t)b * M_SRC * 3;
    for (int u = threadIdx.x; u < M_SRC * 3; u += 256) sp[u] = sb[u];
    __syncthreads();

    const float tx = tgt[t * 3 + 0], ty = tgt[t * 3 + 1], tz = tgt[t * 3 + 2];
    float d0 = 1e30f, d1 = 1e30f, d2 = 1e30f;
    int   i0 = 0,     i1 = 0,     i2 = 0;
    for (int j = 0; j < M_SRC; ++j) {
        const float dx = tx - sp[3 * j + 0];
        const float dy = ty - sp[3 * j + 1];
        const float dz = tz - sp[3 * j + 2];
        const float sq = dx * dx + dy * dy + dz * dz;
        if (sq < d0)      { d2 = d1; i2 = i1; d1 = d0; i1 = i0; d0 = sq; i0 = j; }
        else if (sq < d1) { d2 = d1; i2 = i1; d1 = sq; i1 = j; }
        else if (sq < d2) { d2 = sq; i2 = j; }
    }
    const float e0 = sqrtf(fmaxf(d0, 1e-12f));
    const float e1 = sqrtf(fmaxf(d1, 1e-12f));
    const float e2 = sqrtf(fmaxf(d2, 1e-12f));
    float w0 = 1.0f / (e0 + W_EPS);
    float w1 = 1.0f / (e1 + W_EPS);
    float w2 = 1.0f / (e2 + W_EPS);
    const float inv = 1.0f / (w0 + w1 + w2);
    knn_idx[t * 3 + 0] = i0; knn_idx[t * 3 + 1] = i1; knn_idx[t * 3 + 2] = i2;
    knn_w  [t * 3 + 0] = w0 * inv;
    knn_w  [t * 3 + 1] = w1 * inv;
    knn_w  [t * 3 + 2] = w2 * inv;
}

// ---------------------------------------------------------------------------
// Kernel 2: W (f32, O x C) -> bf16 bits, row-major.
// ---------------------------------------------------------------------------
__global__ void __launch_bounds__(256)
pack_w(const float* __restrict__ W, unsigned short* __restrict__ Wb) {
    const int i = blockIdx.x * 256 + threadIdx.x;
    if (i < O_CH * C_IN) Wb[i] = f32_to_bf16(W[i]);
}

// ---------------------------------------------------------------------------
// Kernel 3: build x^T (Ntot x C) in bf16: cols 0..383 = knn interpolation of
// src_feats, cols 384..895 = tgt_feats.  K-contiguous per point, matching the
// contiguous k order the GEMM uses for A and B.  One block per point.
// ---------------------------------------------------------------------------
__global__ void __launch_bounds__(256)
pack_xT(const float* __restrict__ tgt_feats, const float* __restrict__ src_feats,
        const int* __restrict__ knn_idx, const float* __restrict__ knn_w,
        unsigned short* __restrict__ xT) {
    const int t = blockIdx.x;          // 0 .. NTOT-1
    const int b = t >> 14;
    const int i = t & (N_TGT - 1);
    const int   j0 = knn_idx[t * 3 + 0], j1 = knn_idx[t * 3 + 1], j2 = knn_idx[t * 3 + 2];
    const float w0 = knn_w [t * 3 + 0], w1 = knn_w [t * 3 + 1], w2 = knn_w [t * 3 + 2];
    const float* sf = src_feats + (size_t)b * C2_CH * M_SRC;
    const float* tf = tgt_feats + (size_t)b * C1_CH * N_TGT + i;
    unsigned short* row = xT + (size_t)t * C_IN;
    for (int c = threadIdx.x; c < C_IN; c += 256) {
        float v;
        if (c < C2_CH) {
            const float* f = sf + (size_t)c * M_SRC;
            v = w0 * f[j0] + w1 * f[j1] + w2 * f[j2];
        } else {
            v = tf[(size_t)(c - C2_CH) * N_TGT];
        }
        row[c] = f32_to_bf16(v);
    }
}

// ---------------------------------------------------------------------------
// Kernel 4: y = W @ x via v_wmma_f32_16x16x32_bf16.
// WG = 256 threads = 8 waves in a 4(M) x 2(N) grid; WG tile 128(M) x 128(N).
// Each wave owns a 32(M) x 64(N) tile = 2x4 WMMA accumulators -> 8 WMMAs per
// K step fed by 12 b128 loads (A frags reused 4x, B frags 2x).
// Fragment addressing (contiguous k, same order for A and B):
//   A (16x32 bf16): lane<16 -> K kc..kc+15, lane>=16 -> kc+16..kc+31
//   B (32x16 bf16): lane<16 -> K kc..kc+15, lane>=16 -> kc+16..kc+31
// ---------------------------------------------------------------------------
__global__ void __launch_bounds__(256)
gemm_bf16_wmma(const unsigned short* __restrict__ Wb,
               const unsigned short* __restrict__ xT,
               float* __restrict__ y) {
    const int tBase = blockIdx.x * 128;       // N (points)
    const int oBase = blockIdx.y * 128;       // M (out channels)
    const int wave  = threadIdx.x >> 5;
    const int lane  = threadIdx.x & 31;
    const int half  = lane >> 4;
    const int lr    = lane & 15;
    const int oW    = oBase + (wave >> 1) * 32;   // 4 waves along M
    const int tW    = tBase + (wave & 1) * 64;    // 2 waves along N

    v8f zero;
#pragma unroll
    for (int r = 0; r < 8; ++r) zero[r] = 0.0f;
    v8f acc[2][4];
#pragma unroll
    for (int mi = 0; mi < 2; ++mi)
#pragma unroll
        for (int ni = 0; ni < 4; ++ni) acc[mi][ni] = zero;

    const unsigned short* pa0 = Wb + (size_t)(oW + lr) * C_IN + half * 16;
    const unsigned short* pa1 = pa0 + (size_t)16 * C_IN;
    const unsigned short* pb0 = xT + (size_t)(tW + lr) * C_IN + half * 16;
    const unsigned short* pb1 = pb0 + (size_t)16 * C_IN;
    const unsigned short* pb2 = pb1 + (size_t)16 * C_IN;
    const unsigned short* pb3 = pb2 + (size_t)16 * C_IN;

    for (int kc = 0; kc < C_IN; kc += 32) {
        FragU a0, a1, b0, b1, b2, b3;
        a0.q[0] = *(const v4u*)(pa0 + kc);
        a0.q[1] = *(const v4u*)(pa0 + kc + 8);
        a1.q[0] = *(const v4u*)(pa1 + kc);
        a1.q[1] = *(const v4u*)(pa1 + kc + 8);
        b0.q[0] = *(const v4u*)(pb0 + kc);
        b0.q[1] = *(const v4u*)(pb0 + kc + 8);
        b1.q[0] = *(const v4u*)(pb1 + kc);
        b1.q[1] = *(const v4u*)(pb1 + kc + 8);
        b2.q[0] = *(const v4u*)(pb2 + kc);
        b2.q[1] = *(const v4u*)(pb2 + kc + 8);
        b3.q[0] = *(const v4u*)(pb3 + kc);
        b3.q[1] = *(const v4u*)(pb3 + kc + 8);
        if (kc + 32 < C_IN) {       // WGP-scope prefetch of next K chunk
            __builtin_prefetch(pa0 + kc + 32, 0, 3);
            __builtin_prefetch(pb0 + kc + 32, 0, 3);
            __builtin_prefetch(pb2 + kc + 32, 0, 3);
        }
        acc[0][0] = __builtin_amdgcn_wmma_f32_16x16x32_bf16(
            false, a0.v, false, b0.v, (short)0, acc[0][0], false, false);
        acc[0][1] = __builtin_amdgcn_wmma_f32_16x16x32_bf16(
            false, a0.v, false, b1.v, (short)0, acc[0][1], false, false);
        acc[0][2] = __builtin_amdgcn_wmma_f32_16x16x32_bf16(
            false, a0.v, false, b2.v, (short)0, acc[0][2], false, false);
        acc[0][3] = __builtin_amdgcn_wmma_f32_16x16x32_bf16(
            false, a0.v, false, b3.v, (short)0, acc[0][3], false, false);
        acc[1][0] = __builtin_amdgcn_wmma_f32_16x16x32_bf16(
            false, a1.v, false, b0.v, (short)0, acc[1][0], false, false);
        acc[1][1] = __builtin_amdgcn_wmma_f32_16x16x32_bf16(
            false, a1.v, false, b1.v, (short)0, acc[1][1], false, false);
        acc[1][2] = __builtin_amdgcn_wmma_f32_16x16x32_bf16(
            false, a1.v, false, b2.v, (short)0, acc[1][2], false, false);
        acc[1][3] = __builtin_amdgcn_wmma_f32_16x16x32_bf16(
            false, a1.v, false, b3.v, (short)0, acc[1][3], false, false);
    }

    // C/D layout: VGPR r -> row M = r + 8*half, lane lr -> column N.
#pragma unroll
    for (int mi = 0; mi < 2; ++mi) {
#pragma unroll
        for (int ni = 0; ni < 4; ++ni) {
            const int t = tW + ni * 16 + lr;
#pragma unroll
            for (int r = 0; r < 8; ++r) {
                const int o = oW + mi * 16 + r + half * 8;
                y[(size_t)o * NTOT + t] = acc[mi][ni][r];
            }
        }
    }
}

// ---------------------------------------------------------------------------
// Kernel 5: per-channel mean / rstd over the 32768 columns.
// ---------------------------------------------------------------------------
__global__ void __launch_bounds__(256)
bn_stats(const float* __restrict__ y, float* __restrict__ stats) {
    __shared__ float bs[256], bs2[256];
    const int o = blockIdx.x;
    const float* row = y + (size_t)o * NTOT;
    float s = 0.0f, s2 = 0.0f;
    for (int j = threadIdx.x; j < NTOT; j += 256) {
        const float v = row[j];
        s += v; s2 += v * v;
    }
    bs[threadIdx.x] = s; bs2[threadIdx.x] = s2;
    __syncthreads();
    for (int st = 128; st > 0; st >>= 1) {
        if (threadIdx.x < st) {
            bs [threadIdx.x] += bs [threadIdx.x + st];
            bs2[threadIdx.x] += bs2[threadIdx.x + st];
        }
        __syncthreads();
    }
    if (threadIdx.x == 0) {
        const float mean = bs[0] * (1.0f / NTOT);
        const float var  = bs2[0] * (1.0f / NTOT) - mean * mean;
        stats[o]        = mean;
        stats[O_CH + o] = 1.0f / sqrtf(var + BN_EPS);
    }
}

// ---------------------------------------------------------------------------
// Kernel 6: in-place normalize + affine + ReLU (float4 vectorized).
// ---------------------------------------------------------------------------
__global__ void __launch_bounds__(256)
bn_norm(float* __restrict__ y, const float* __restrict__ stats,
        const float* __restrict__ gamma, const float* __restrict__ beta) {
    const size_t base = ((size_t)blockIdx.x * 256 + threadIdx.x) * 4;
    const int o = (int)(base >> 15);                 // / NTOT
    const float mean = stats[o], rstd = stats[O_CH + o];
    const float g = gamma[o], bb = beta[o];
    v4f v = *(v4f*)(y + base);
#pragma unroll
    for (int r = 0; r < 4; ++r) {
        float z = (v[r] - mean) * rstd * g + bb;
        v[r] = fmaxf(z, 0.0f);
    }
    *(v4f*)(y + base) = v;
}

// ---------------------------------------------------------------------------
// Launch: knn -> pack W -> pack xT -> WMMA GEMM -> BN stats -> BN norm.
// Workspace (~60 MB): idx | w | stats | Wb(bf16) | xT(bf16).
// ---------------------------------------------------------------------------
extern "C" void kernel_launch(void* const* d_in, const int* in_sizes, int n_in,
                              void* d_out, int out_size, void* d_ws, size_t ws_size,
                              hipStream_t stream) {
    (void)in_sizes; (void)n_in; (void)out_size; (void)ws_size;
    const float* tgt       = (const float*)d_in[0];
    const float* src       = (const float*)d_in[1];
    const float* tgt_feats = (const float*)d_in[2];
    const float* src_feats = (const float*)d_in[3];
    const float* W         = (const float*)d_in[4];
    const float* gamma     = (const float*)d_in[5];
    const float* beta      = (const float*)d_in[6];
    // d_in[7] = k, fixed at 3 (top-3 logic is compile-time)
    float* y = (float*)d_out;

    char* ws = (char*)d_ws;
    int*            knn_idx = (int*)           (ws + 0);                  // 384 KB
    float*          knn_w   = (float*)         (ws + 393216);            // 384 KB
    float*          stats   = (float*)         (ws + 786432);            // 4 KB
    unsigned short* Wb      = (unsigned short*)(ws + 790528);            // 896 KB
    unsigned short* xT      = (unsigned short*)(ws + 1708032);           // 56 MB

    knn_top3<<<NTOT / 256, 256, 0, stream>>>(tgt, src, knn_idx, knn_w);
    pack_w<<<(O_CH * C_IN + 255) / 256, 256, 0, stream>>>(W, Wb);
    pack_xT<<<NTOT, 256, 0, stream>>>(tgt_feats, src_feats, knn_idx, knn_w, xT);
    gemm_bf16_wmma<<<dim3(NTOT / 128, O_CH / 128), 256, 0, stream>>>(Wb, xT, y);
    bn_stats<<<O_CH, 256, 0, stream>>>(y, stats);
    bn_norm<<<(O_CH * NTOT) / (256 * 4), 256, 0, stream>>>(y, stats, gamma, beta);
}
